// TypeNet_7421703488259
// MI455X (gfx1250) — compile-verified, matrix operand order checked by reading, not run on hardware
//
#include <hip/hip_runtime.h>
#include <stdint.h>

#define H  128
#define TT 128
#define BB 512

typedef __attribute__((ext_vector_type(16))) _Float16 v16h;
typedef __attribute__((ext_vector_type(8)))  float    v8f;
typedef __attribute__((ext_vector_type(4)))  unsigned int u32x4;
typedef __attribute__((ext_vector_type(8)))  int i32x8;
typedef __attribute__((ext_vector_type(4)))  int i32x4;

#ifndef __has_builtin
#define __has_builtin(x) 0
#endif
#if __has_builtin(__builtin_amdgcn_tensor_load_to_lds) && __has_builtin(__builtin_amdgcn_s_wait_tensorcnt)
#define HAVE_TDM 1
#else
#define HAVE_TDM 0
#endif

#if HAVE_TDM
// Build a TDM D# for a 2D tile load (tile0 elems/row, tile1 rows, row stride
// stride0 elems, element size 2^ds_log2 bytes) per CDNA5 ISA.
// This toolchain's builtin takes 6 args: g0, g1, g2, g3, g4(spare), cpol.
__device__ __forceinline__ void tdm_load_2d(uint32_t lds_off, const void* gptr,
    uint32_t ds_log2, uint32_t tdim0, uint32_t tdim1,
    uint32_t tile0, uint32_t tile1, uint64_t stride0) {
  uint64_t ga = (uint64_t)(uintptr_t)gptr;
  u32x4 g0;
  g0.x = 1u;                                        // count=1, user mode
  g0.y = lds_off;                                   // lds_addr
  g0.z = (uint32_t)ga;                              // global_addr[31:0]
  g0.w = (uint32_t)((ga >> 32) & 0x01FFFFFFu) | (2u << 30);  // addr hi | type=2
  i32x8 g1;
  g1[0] = (int)(ds_log2 << 16);                     // wg_mask=0, data_size
  g1[1] = (int)((tdim0 & 0xFFFFu) << 16);           // tensor_dim0[15:0]
  g1[2] = (int)((tdim0 >> 16) | ((tdim1 & 0xFFFFu) << 16));
  g1[3] = (int)((tdim1 >> 16) | (tile0 << 16));     // tile_dim0
  g1[4] = (int)(tile1 & 0xFFFFu);                   // tile_dim1, tile_dim2=0
  g1[5] = (int)(uint32_t)stride0;                   // tensor_dim0_stride lo
  g1[6] = (int)(uint32_t)((stride0 >> 32) & 0xFFFFu);
  g1[7] = 0;
  i32x4 gz4 = {0, 0, 0, 0};
  i32x8 gz8 = {0, 0, 0, 0, 0, 0, 0, 0};
  __builtin_amdgcn_tensor_load_to_lds(g0, g1, gz4, gz4, gz8, 0);
}
#endif

__device__ __forceinline__ uint32_t pk2h(float a, float b) {
  union { _Float16 h; unsigned short u; } ua, ub;
  ua.h = (_Float16)a; ub.h = (_Float16)b;
  return (uint32_t)ua.u | ((uint32_t)ub.u << 16);
}

// Convert W [512 rows][128 cols] (row-major) into f16 B-fragments of W^T
// (K=128 x N=512). Fragment (kc,nt): lane L, word r holds K = kc*32 +
// 16*(r/4) + 8*(L/16) + 2*(r%4) per ISA 16-bit B layout; N = nt*16 + L%16.
__global__ void swizzle_kernel(const float* __restrict__ W, uint32_t* __restrict__ dst) {
  int id = blockIdx.x * blockDim.x + threadIdx.x;
  if (id >= 32768) return;
  int r  = id & 7;
  int L  = (id >> 3) & 31;
  int nt = (id >> 8) & 31;
  int kc = id >> 13;
  int n  = nt * 16 + (L & 15);
  int k0 = kc * 32 + ((r >> 2) << 4) + ((L >> 4) << 3) + ((r & 3) << 1);
  dst[id] = pk2h(W[n * 128 + k0], W[n * 128 + k0 + 1]);
}

// xw1[bt, g] = bih[g] + bhh[g] + sum_d x[bt,d] * Wih1[g,d]   (d < 5)
__global__ void xw1_kernel(const float* __restrict__ x, const float* __restrict__ Wih,
                           const float* __restrict__ bih, const float* __restrict__ bhh,
                           float* __restrict__ xw, int total) {
  int idx = blockIdx.x * blockDim.x + threadIdx.x;
  int stride = gridDim.x * blockDim.x;
  for (int i = idx; i < total; i += stride) {
    int g = i & 511;
    size_t bt = (size_t)(i >> 9);
    const float* xp = x + bt * 5;
    const float* wp = Wih + (size_t)g * 5;
    xw[i] = bih[g] + bhh[g] + xp[0]*wp[0] + xp[1]*wp[1] + xp[2]*wp[2]
                            + xp[3]*wp[3] + xp[4]*wp[4];
  }
}

// Persistent per-batch-tile LSTM. Block = 4 waves, 16 batch rows; wave w owns
// gate w's 128 output columns. TDM streams xW tiles one step ahead; B-matrix
// fragments are software-pipelined (prefetch tile nt+1 during WMMA of nt).
__global__ __launch_bounds__(128) void lstm_kernel(
    const float* __restrict__ xw,        // [B, T, 4H] precomputed input proj + biases
    const uint32_t* __restrict__ whh_sw, // 32768 u32 swizzled f16 Whh^T fragments
    float* __restrict__ out)             // [B, T, H]
{
  extern __shared__ char smem[];
  uint32_t* lds_whh  = (uint32_t*)(smem);                            // 131072 B
  float*    lds_xw   = (float*)(smem + 131072);                      // 2*[16][512]
  float*    lds_gact = (float*)(smem + 131072 + 65536);              // [16][128][4]
  uint32_t* lds_hA   = (uint32_t*)(smem + 131072 + 65536 + 32768);   // [4][32][8]
  _Float16* lds_hA16 = (_Float16*)lds_hA;

  const int tid  = threadIdx.x;
  const int wid  = tid >> 5;
  const int lane = tid & 31;
  const int loL  = lane & 15, hiL = lane >> 4;
  const size_t b0 = (size_t)blockIdx.x * 16;

  for (int i = tid; i < 1024; i += 128) lds_hA[i] = 0u;   // h0 = 0

#if HAVE_TDM
  if (wid == 0) {
    tdm_load_2d((uint32_t)(size_t)lds_whh, whh_sw, 3u, 16384u, 1u, 16384u, 1u, 16384ull);
    tdm_load_2d((uint32_t)(size_t)lds_xw, xw + b0 * TT * 512,
                2u, 512u, 1u << 20, 512u, 16u, (uint64_t)TT * 512);
  }
#else
  for (int i = tid; i < 32768; i += 128) lds_whh[i] = whh_sw[i];
#endif

  float c[16];
#pragma unroll
  for (int j = 0; j < 16; ++j) c[j] = 0.f;

  for (int t = 0; t < TT; ++t) {
#if HAVE_TDM
    if (wid == 0) __builtin_amdgcn_s_wait_tensorcnt(0);
    __syncthreads();   // xw[t] + hA(t-1) visible
    if (wid == 0 && (t + 1) < TT) {
      tdm_load_2d((uint32_t)(size_t)(lds_xw + ((t + 1) & 1) * 8192),
                  xw + (b0 * TT + (size_t)(t + 1)) * 512,
                  2u, 512u, 1u << 20, 512u, 16u, (uint64_t)TT * 512);
    }
    const float* xwrow = lds_xw + (t & 1) * 8192;
#else
    __syncthreads();
    {
      const float* src = xw + (b0 * TT + (size_t)t) * 512;
      for (int i = tid; i < 16 * 512; i += 128) {
        int j = i >> 9, cc = i & 511;
        lds_xw[i] = src[(size_t)j * TT * 512 + cc];
      }
    }
    __syncthreads();
    const float* xwrow = lds_xw;
#endif

    // ---- WMMA phase: g_pre = h @ Whh^T (wave w -> gate w's 128 cols) ----
    v16h afr[4];
#pragma unroll
    for (int kc = 0; kc < 4; ++kc)
      afr[kc] = *(const v16h*)&lds_hA[(kc * 32 + lane) * 8];
    v16h bcur[4], bnxt[4];
#pragma unroll
    for (int kc = 0; kc < 4; ++kc)
      bcur[kc] = *(const v16h*)&lds_whh[((kc * 32 + wid * 8) * 32 + lane) * 8];
#pragma unroll
    for (int nt = 0; nt < 8; ++nt) {
      if (nt < 7) {
#pragma unroll
        for (int kc = 0; kc < 4; ++kc)
          bnxt[kc] = *(const v16h*)&lds_whh[((kc * 32 + wid * 8 + nt + 1) * 32 + lane) * 8];
      }
      v8f acc = {};
#pragma unroll
      for (int kc = 0; kc < 4; ++kc)
        acc = __builtin_amdgcn_wmma_f32_16x16x32_f16(
            false, afr[kc], false, bcur[kc], (short)0, acc, false, false);
#pragma unroll
      for (int r = 0; r < 8; ++r)
        lds_gact[((r + 8 * hiL) * 128 + nt * 16 + loL) * 4 + wid] = acc[r];
#pragma unroll
      for (int kc = 0; kc < 4; ++kc) bcur[kc] = bnxt[kc];
    }
    __syncthreads();   // gates visible; hA reads complete

    // ---- elementwise: thread k owns column k for all 16 rows ----
    const int k = tid;
    const int p = k & 1, rr = ((k >> 4) & 1) * 4 + ((k >> 1) & 3);
    const int hi2 = (k >> 3) & 1, kc2 = k >> 5;
    const float4* g4p = (const float4*)lds_gact;
#pragma unroll
    for (int j = 0; j < 16; ++j) {
      float4 g4 = g4p[j * 128 + k];            // (i, f, g, o) pre-activations
      float ig = g4.x + xwrow[j * 512 +       k];
      float fg = g4.y + xwrow[j * 512 + 128 + k];
      float gg = g4.z + xwrow[j * 512 + 256 + k];
      float og = g4.w + xwrow[j * 512 + 384 + k];
      float iv = 1.f / (1.f + __expf(-ig));
      float fv = 1.f / (1.f + __expf(-fg));
      float gv = tanhf(gg);
      float ov = 1.f / (1.f + __expf(-og));
      c[j] = fv * c[j] + iv * gv;
      float hv = ov * tanhf(c[j]);
      out[((b0 + j) * TT + t) * 128 + k] = hv;
      lds_hA16[((kc2 * 32 + hi2 * 16 + j) * 8 + rr) * 2 + p] = (_Float16)hv;
    }
  }
}

// Per-channel BN stats over (B,T): scale = gamma*rsqrt(var+eps), shift = beta - mean*scale
__global__ void bnstats_kernel(const float* __restrict__ x, const float* __restrict__ gamma,
                               const float* __restrict__ beta, float* __restrict__ scale,
                               float* __restrict__ shift) {
  __shared__ float s1[256], s2[256];
  int c = blockIdx.x, tid = threadIdx.x;
  const int N = BB * TT;
  float a = 0.f, b = 0.f;
  for (int i = tid; i < N; i += 256) {
    float v = x[(size_t)i * 128 + c];
    a += v; b += v * v;
  }
  s1[tid] = a; s2[tid] = b; __syncthreads();
  for (int off = 128; off > 0; off >>= 1) {
    if (tid < off) { s1[tid] += s1[tid + off]; s2[tid] += s2[tid + off]; }
    __syncthreads();
  }
  if (tid == 0) {
    float mean = s1[0] / (float)N;
    float var  = s2[0] / (float)N - mean * mean;
    float rs = rsqrtf(var + 1e-5f);
    float sc = gamma[c] * rs;
    scale[c] = sc;
    shift[c] = beta[c] - mean * sc;
  }
}

__global__ void bnapply_kernel(float* __restrict__ x, const float* __restrict__ scale,
                               const float* __restrict__ shift, int total) {
  int idx = blockIdx.x * blockDim.x + threadIdx.x;
  int stride = gridDim.x * blockDim.x;
  for (int i = idx; i < total; i += stride) {
    int ch = i & 127;
    x[i] = x[i] * scale[ch] + shift[ch];
  }
}

// xw2[m, :] = bn_out[m, :] @ Wih2^T + (bih2 + bhh2), WMMA, M = B*T.
__global__ __launch_bounds__(128) void gemm_xw2_kernel(
    const float* __restrict__ bnout, const uint32_t* __restrict__ w_sw,
    const float* __restrict__ bih, const float* __restrict__ bhh,
    float* __restrict__ xw2, int Mtiles) {
  extern __shared__ char smem[];
  uint32_t* lds_w  = (uint32_t*)smem;                 // 131072 B
  uint32_t* lds_hA = (uint32_t*)(smem + 131072);      // 4096 B
  _Float16* lds_hA16 = (_Float16*)lds_hA;
  const int tid = threadIdx.x, wid = tid >> 5, lane = tid & 31;
  const int loL = lane & 15, hiL = lane >> 4;

#if HAVE_TDM
  if (wid == 0) {
    tdm_load_2d((uint32_t)(size_t)lds_w, w_sw, 3u, 16384u, 1u, 16384u, 1u, 16384ull);
    __builtin_amdgcn_s_wait_tensorcnt(0);
  }
#else
  for (int i = tid; i < 32768; i += 128) lds_w[i] = w_sw[i];
#endif
  float bias[8];
#pragma unroll
  for (int nt = 0; nt < 8; ++nt) {
    int col = wid * 128 + nt * 16 + loL;
    bias[nt] = bih[col] + bhh[col];
  }
  __syncthreads();

  const int k = tid;
  const int p = k & 1, rr = ((k >> 4) & 1) * 4 + ((k >> 1) & 3);
  const int hi2 = (k >> 3) & 1, kc2 = k >> 5;

  for (int mt = blockIdx.x; mt < Mtiles; mt += gridDim.x) {
    size_t m0 = (size_t)mt * 16;
#pragma unroll
    for (int j = 0; j < 16; ++j) {
      float v = bnout[(m0 + j) * 128 + k];
      lds_hA16[((kc2 * 32 + hi2 * 16 + j) * 8 + rr) * 2 + p] = (_Float16)v;
    }
    __syncthreads();
    v16h afr[4];
#pragma unroll
    for (int kc = 0; kc < 4; ++kc)
      afr[kc] = *(const v16h*)&lds_hA[(kc * 32 + lane) * 8];
    v16h bcur[4], bnxt[4];
#pragma unroll
    for (int kc = 0; kc < 4; ++kc)
      bcur[kc] = *(const v16h*)&lds_w[((kc * 32 + wid * 8) * 32 + lane) * 8];
#pragma unroll
    for (int nt = 0; nt < 8; ++nt) {
      if (nt < 7) {
#pragma unroll
        for (int kc = 0; kc < 4; ++kc)
          bnxt[kc] = *(const v16h*)&lds_w[((kc * 32 + wid * 8 + nt + 1) * 32 + lane) * 8];
      }
      v8f acc = {};
#pragma unroll
      for (int kc = 0; kc < 4; ++kc)
        acc = __builtin_amdgcn_wmma_f32_16x16x32_f16(
            false, afr[kc], false, bcur[kc], (short)0, acc, false, false);
      int col = wid * 128 + nt * 16 + loL;
#pragma unroll
      for (int r = 0; r < 8; ++r)
        xw2[(m0 + (size_t)(r + 8 * hiL)) * 512 + col] = acc[r] + bias[nt];
#pragma unroll
      for (int kc = 0; kc < 4; ++kc) bcur[kc] = bnxt[kc];
    }
    __syncthreads();
  }
}

// FC + (folded layer-2 BN) + L2 normalize. Block = batch row, thread = output.
__global__ __launch_bounds__(128) void fc_kernel(
    const float* __restrict__ h2, const float* __restrict__ scale,
    const float* __restrict__ shift, const float* __restrict__ fcW,
    const float* __restrict__ fcb, float* __restrict__ outp) {
  __shared__ float xn[128];
  __shared__ float red[128];
  int b = blockIdx.x, o = threadIdx.x;
  float xv = h2[((size_t)b * TT + (TT - 1)) * 128 + o];
  xn[o] = xv * scale[o] + shift[o];
  __syncthreads();
  float s = fcb[o];
  for (int hh = 0; hh < 128; ++hh) s += fcW[o * 128 + hh] * xn[hh];
  red[o] = s * s; __syncthreads();
  for (int off = 64; off > 0; off >>= 1) {
    if (o < off) red[o] += red[o + off];
    __syncthreads();
  }
  float nrm = sqrtf(red[0]);
  outp[(size_t)b * 128 + o] = s / fmaxf(nrm, 1e-12f);
}

extern "C" void kernel_launch(void* const* d_in, const int* in_sizes, int n_in,
                              void* d_out, int out_size, void* d_ws, size_t ws_size,
                              hipStream_t stream) {
  const float* A    = (const float*)d_in[0];
  const float* P    = (const float*)d_in[1];
  const float* Nn   = (const float*)d_in[2];
  const float* Wih1 = (const float*)d_in[3];
  const float* Whh1 = (const float*)d_in[4];
  const float* bih1 = (const float*)d_in[5];
  const float* bhh1 = (const float*)d_in[6];
  const float* g1   = (const float*)d_in[7];
  const float* b1   = (const float*)d_in[8];
  const float* Wih2 = (const float*)d_in[9];
  const float* Whh2 = (const float*)d_in[10];
  const float* bih2 = (const float*)d_in[11];
  const float* bhh2 = (const float*)d_in[12];
  const float* g2   = (const float*)d_in[13];
  const float* b2   = (const float*)d_in[14];
  const float* fcW  = (const float*)d_in[15];
  const float* fcb  = (const float*)d_in[16];
  float* out = (float*)d_out;

  char* ws = (char*)d_ws;
  float*    xwbuf = (float*)ws;                               // 128 MB [B,T,512]
  float*    hbuf  = (float*)(ws + 134217728);                 //  32 MB [B,T,128]
  uint32_t* whh1s = (uint32_t*)(ws + 134217728 + 33554432);   // 128 KB each
  uint32_t* whh2s = whh1s + 32768;
  uint32_t* wih2s = whh2s + 32768;
  float*    scale = (float*)(wih2s + 32768);
  float*    shift = scale + 128;

  (void)hipFuncSetAttribute((const void*)lstm_kernel,
      hipFuncAttributeMaxDynamicSharedMemorySize, 233472);
  (void)hipFuncSetAttribute((const void*)gemm_xw2_kernel,
      hipFuncAttributeMaxDynamicSharedMemorySize, 135168);

  swizzle_kernel<<<128, 256, 0, stream>>>(Whh1, whh1s);
  swizzle_kernel<<<128, 256, 0, stream>>>(Whh2, whh2s);
  swizzle_kernel<<<128, 256, 0, stream>>>(Wih2, wih2s);

  const float* ins[3] = {A, P, Nn};
  for (int i = 0; i < 3; ++i) {
    xw1_kernel<<<2048, 256, 0, stream>>>(ins[i], Wih1, bih1, bhh1, xwbuf, 512 * 128 * 512);
    lstm_kernel<<<32, 128, 233472, stream>>>(xwbuf, whh1s, hbuf);
    bnstats_kernel<<<128, 256, 0, stream>>>(hbuf, g1, b1, scale, shift);
    bnapply_kernel<<<2048, 256, 0, stream>>>(hbuf, scale, shift, 512 * 128 * 128);
    gemm_xw2_kernel<<<256, 128, 135168, stream>>>(hbuf, wih2s, bih2, bhh2, xwbuf, 4096);
    lstm_kernel<<<32, 128, 233472, stream>>>(xwbuf, whh2s, hbuf);
    bnstats_kernel<<<128, 256, 0, stream>>>(hbuf, g2, b2, scale, shift);
    fc_kernel<<<512, 128, 0, stream>>>(hbuf, scale, shift, fcW, fcb,
                                       out + (size_t)i * 512 * 128);
  }
}
